// StaticSparseCRF_46746424050045
// MI455X (gfx1250) — compile-verified, matrix operand order checked by reading, not run on hardware
//
#include <hip/hip_runtime.h>
#include <cstdint>
#include <cstddef>

// Problem constants (fixed by the reference).
#define KSTATES 529
#define TLEN    2048
#define NPAD    544           // 529 padded to 34*16 (and 17*32)
#define NT      34            // N tiles of 16
#define KKT     17            // reduce-dim tiles of 32
#define FORBID  (-10000.0f)

typedef _Float16 f16;
typedef _Float16 v16h __attribute__((ext_vector_type(16)));
typedef _Float16 v8h  __attribute__((ext_vector_type(8)));
typedef float    v8f  __attribute__((ext_vector_type(8)));
typedef float    v4f  __attribute__((ext_vector_type(4)));
typedef int      g4i  __attribute__((vector_size(16)));   // matches builtin sig

union Frag16 { v16h v; v8h h[2]; };

#define AS1 __attribute__((address_space(1)))
#define AS3 __attribute__((address_space(3)))

#if defined(__has_builtin)
#if __has_builtin(__builtin_amdgcn_global_load_async_to_lds_b128) && \
    __has_builtin(__builtin_amdgcn_global_load_async_to_lds_b32)
#define HAVE_ASYNC_LDS 1
#endif
#endif
#ifndef HAVE_ASYNC_LDS
#define HAVE_ASYNC_LDS 0
#endif

__device__ __forceinline__ void wait_async0() {
#if HAVE_ASYNC_LDS
#if __has_builtin(__builtin_amdgcn_s_wait_asynccnt)
  __builtin_amdgcn_s_wait_asynccnt(0);
#else
  asm volatile("s_wait_asynccnt 0x0" ::: "memory");
#endif
#endif
}

// Raw-rate transcendental helpers (v_exp_f32 / v_log_f32).
__device__ __forceinline__ float fast_exp(float x) {
#if defined(__has_builtin) && __has_builtin(__builtin_amdgcn_exp2f)
  return __builtin_amdgcn_exp2f(x * 1.44269504088896340736f);
#else
  return __expf(x);
#endif
}
__device__ __forceinline__ float fast_log(float x) {
#if defined(__has_builtin) && __has_builtin(__builtin_amdgcn_logf)
  return __builtin_amdgcn_logf(x) * 0.69314718055994530942f;
#else
  return __logf(x);
#endif
}

// Copy one step's (16 x 529 f32) emissions tile into an LDS buffer.
// Row r = tid>>6 handled by 64 threads: 132 x b128 chunks + 1 x b32 tail,
// via the CDNA5 async global->LDS engine (ASYNCcnt) when available.
__device__ __forceinline__ void fill_emit(const float* __restrict__ em,
                                          float (*dst)[NPAD],
                                          int b0, int B, int t, int tid) {
  const int r  = tid >> 6;
  const int q0 = tid & 63;
  const int b  = b0 + r;
  if (b >= B) return;
  const float* grow = &em[((size_t)b * TLEN + t) * KSTATES];
  for (int q = q0; q < 133; q += 64) {
    const int col = (q < 132) ? (q << 2) : 528;
    const float* g = grow + col;
    float* l = &dst[r][col];
#if HAVE_ASYNC_LDS
    if (q < 132)
      __builtin_amdgcn_global_load_async_to_lds_b128(
          (AS1 g4i*)(uintptr_t)g, (AS3 g4i*)(uint32_t)(uintptr_t)l, 0, 0);
    else
      __builtin_amdgcn_global_load_async_to_lds_b32(
          (AS1 int*)(uintptr_t)g, (AS3 int*)(uint32_t)(uintptr_t)l, 0, 0);
#else
    if (q < 132) *(v4f*)l = *(const v4f*)g;
    else         *l = *g;
#endif
  }
}

// ---------------------------------------------------------------------------
// Zero the swizzled-W buffer + logZ/gold slots in workspace.
// ---------------------------------------------------------------------------
__global__ void zero_ws_kernel(uint32_t* __restrict__ p, int n) {
  for (int i = blockIdx.x * blockDim.x + threadIdx.x; i < n;
       i += gridDim.x * blockDim.x)
    p[i] = 0u;
}

// ---------------------------------------------------------------------------
// Build W[i][j] = exp(transitions[j][m]) where i = inc_idx[j][m] >= 0,
// stored directly in the per-lane v_wmma_f32_16x16x32_f16 B-fragment layout:
//   tile (nt,kk) -> 32 lanes x 16 halves, contiguous 1KB blocks.
// ---------------------------------------------------------------------------
__global__ void build_w_kernel(const float* __restrict__ trans,
                               const int*   __restrict__ inc_idx,
                               f16* __restrict__ wsw, int M) {
  int e = blockIdx.x * blockDim.x + threadIdx.x;
  if (e >= KSTATES * M) return;
  int j = e / M;
  int i = inc_idx[e];
  if (i < 0) return;                       // inc_mask == false slots
  float w = __expf(trans[e]);
  int nt = j >> 4, ncol = j & 15;
  int kk = i >> 5, krel = i & 31;
  int lane = ncol + (((krel >> 3) & 1) << 4);
  int h    = (((krel >> 4) & 1) << 3) + (krel & 7);
  size_t addr = ((size_t)(nt * KKT + kk) * 32 + lane) * 16 + h;
  wsw[addr] = (f16)w;
}

// ---------------------------------------------------------------------------
// Forward scan. One workgroup per 16 batch rows (one WMMA M-tile).
// Per step: (16 x 544) x (544 x 544) f16 WMMA GEMM + log/emit epilogue.
// Emissions double-buffered in LDS via async global->LDS copies; the
// sequence mask slab is LDS-resident for the whole scan.
// ---------------------------------------------------------------------------
__global__ __launch_bounds__(1024)
void crf_forward_kernel(const float* __restrict__ em,
                        const unsigned char* __restrict__ seqmask,
                        const f16* __restrict__ wsw,
                        float* __restrict__ logZ, int B) {
  __shared__ float s_alpha[16][NPAD];                 // 34.0 KB
  __shared__ __align__(16) f16 s_aexp[16][NPAD];      // 17.0 KB
  __shared__ __align__(16) float s_emit[2][16][NPAD]; // 69.6 KB
  __shared__ uint32_t s_mask32[16 * (TLEN / 4)];      // 32.0 KB
  __shared__ float s_amax[16];
  __shared__ float s_sum[16];

  const int tid  = threadIdx.x;
  const int lane = tid & 31;
  const int wave = tid >> 5;                          // 0..31
  const int b0   = blockIdx.x * 16;
  const int m    = tid >> 6;                          // batch row 0..15
  const int k0   = tid & 63;                          // state phase

  // alpha_0 = emissions[:, 0, :]
  {
    const int b = b0 + m;
    for (int k = k0; k < KSTATES; k += 64)
      s_alpha[m][k] = (b < B) ? em[((size_t)b * TLEN + 0) * KSTATES + k]
                              : -3.0e38f;
  }
  // K-dim padding of the A operand contributes 0 forever.
  if (tid < 16)
    for (int k = KSTATES; k < NPAD; ++k) s_aexp[tid][k] = (f16)0.0f;

  // Stage the whole mask slab (16 rows x 2048 bytes) into LDS once.
  {
    const uint32_t* gm = (const uint32_t*)seqmask;
    for (int w = tid; w < 16 * (TLEN / 4); w += 1024) {
      const int r = w / (TLEN / 4);
      const int b = b0 + r;
      s_mask32[w] = (b < B) ? gm[(size_t)b * (TLEN / 4) + (w % (TLEN / 4))]
                            : 0u;
    }
  }
  const uint8_t* s_mask8 = (const uint8_t*)s_mask32;

  // Pre-stage emissions for t=1.
  fill_emit(em, s_emit[1], b0, B, 1, tid);
  wait_async0();

  const v8h* W8 = (const v8h*)wsw;

  for (int t = 1; t < TLEN; ++t) {
    __syncthreads();                                  // alpha + emit staged
    if (tid < 16) s_amax[tid] = -3.0e38f;
    __syncthreads();

    // Phase 1: per-batch row max; L2 prefetch two steps ahead.
    {
      float lm = -3.0e38f;
      const int b = b0 + m;
      for (int k = k0; k < KSTATES; k += 64) {
        lm = fmaxf(lm, s_alpha[m][k]);
        if (((k & 31) == 0) && (b < B) && (t + 2) < TLEN)
          __builtin_prefetch(&em[((size_t)b * TLEN + (t + 2)) * KSTATES + k],
                             0, 3);                   // global_prefetch_b8
      }
      atomicMax(&s_amax[m], lm);                      // ds float max
    }
    __syncthreads();

    // Phase 2: A operand = exp(alpha - amax) in f16.
    {
      const float am = s_amax[m];
      for (int k = k0; k < KSTATES; k += 64)
        s_aexp[m][k] = (f16)fast_exp(s_alpha[m][k] - am);
    }
    __syncthreads();

    // Phase 3: kick off async emissions copy for t+1, then WMMA GEMM.
    if (t + 1 < TLEN) fill_emit(em, s_emit[(t + 1) & 1], b0, B, t + 1, tid);

    const float(*emt)[NPAD] = s_emit[t & 1];
    const int hi8  = (lane >> 4) << 3;
    const int arow = lane & 15;
#pragma unroll 1
    for (int nt = wave; nt < NT; nt += 32) {
      v8f acc = {};
      const f16* ap = &s_aexp[arow][hi8];
      const v8h* wp = W8 + (size_t)nt * KKT * 64 + lane * 2;
#pragma unroll 1
      for (int kk = 0; kk < KKT; ++kk) {
        Frag16 a, bf;
        a.h[0] = *(const v8h*)(ap);                   // ds_load_b128
        a.h[1] = *(const v8h*)(ap + 16);              // ds_load_b128
        bf.h[0] = wp[0];                              // global_load_b128
        bf.h[1] = wp[1];
        acc = __builtin_amdgcn_wmma_f32_16x16x32_f16(
            false, a.v, false, bf.v, (short)0, acc, false, false);
        ap += 32;
        wp += 64;
      }
      const int n = (nt << 4) + (lane & 15);
      if (n < KSTATES) {
#pragma unroll
        for (int r = 0; r < 8; ++r) {
          const int mrow = ((lane >> 4) << 3) + r;    // C/D row layout
          const float nv = fast_log(acc[r]) + s_amax[mrow] + emt[mrow][n];
          if (s_mask8[mrow * TLEN + t])               // ds_load_u8
            s_alpha[mrow][n] = nv;
        }
      }
    }
    wait_async0();   // t+1 tile resident before next iteration's barrier
  }

  // logZ[b] = amax + log(sum exp(alpha - amax))
  __syncthreads();
  if (tid < 16) { s_amax[tid] = -3.0e38f; s_sum[tid] = 0.0f; }
  __syncthreads();
  {
    float lm = -3.0e38f;
    for (int k = k0; k < KSTATES; k += 64) lm = fmaxf(lm, s_alpha[m][k]);
    atomicMax(&s_amax[m], lm);
  }
  __syncthreads();
  {
    const float am = s_amax[m];
    float ls = 0.0f;
    for (int k = k0; k < KSTATES; k += 64) ls += fast_exp(s_alpha[m][k] - am);
    atomicAdd(&s_sum[m], ls);                        // ds_add_f32
  }
  __syncthreads();
  if (tid < 16 && (b0 + tid) < B)
    logZ[b0 + tid] = s_amax[tid] + fast_log(s_sum[tid]);
}

// ---------------------------------------------------------------------------
// Gold path score (embarrassingly parallel).
// ---------------------------------------------------------------------------
__global__ void gold_kernel(const float* __restrict__ em,
                            const int* __restrict__ tags,
                            const unsigned char* __restrict__ seqmask,
                            const float* __restrict__ trans,
                            const int* __restrict__ inc_idx,
                            float* __restrict__ gold, int M) {
  __shared__ float red[256];
  const int b = blockIdx.x;
  const int tid = threadIdx.x;
  float acc = 0.0f;
  for (int t = 1 + tid; t < TLEN; t += 256) {
    const int pt = tags[(size_t)b * TLEN + t - 1];
    const int ct = tags[(size_t)b * TLEN + t];
    float tr = FORBID;
    for (int mm = 0; mm < M; ++mm)
      if (inc_idx[(size_t)ct * M + mm] == pt)
        tr = fmaxf(tr, trans[(size_t)ct * M + mm]);
    const float e  = em[((size_t)b * TLEN + t) * KSTATES + ct];
    const float mf = seqmask[(size_t)b * TLEN + t] ? 1.0f : 0.0f;
    acc += (e + tr) * mf;
  }
  red[tid] = acc;
  __syncthreads();
  for (int s = 128; s > 0; s >>= 1) {
    if (tid < s) red[tid] += red[tid + s];
    __syncthreads();
  }
  if (tid == 0) {
    const int t0 = tags[(size_t)b * TLEN];
    gold[b] = red[0] + em[(size_t)b * TLEN * KSTATES + t0];
  }
}

// ---------------------------------------------------------------------------
// out = mean(logZ - gold)
// ---------------------------------------------------------------------------
__global__ void final_kernel(const float* __restrict__ logZ,
                             const float* __restrict__ gold,
                             float* __restrict__ out, int B) {
  __shared__ float red[64];
  const int tid = threadIdx.x;
  red[tid] = (tid < B) ? (logZ[tid] - gold[tid]) : 0.0f;
  __syncthreads();
  for (int s = 32; s > 0; s >>= 1) {
    if (tid < s) red[tid] += red[tid + s];
    __syncthreads();
  }
  if (tid == 0) out[0] = red[0] / (float)B;
}

// ---------------------------------------------------------------------------
extern "C" void kernel_launch(void* const* d_in, const int* in_sizes, int n_in,
                              void* d_out, int out_size, void* d_ws,
                              size_t ws_size, hipStream_t stream) {
  const float* em            = (const float*)d_in[0];
  const int* tags            = (const int*)d_in[1];
  const unsigned char* mask  = (const unsigned char*)d_in[2];
  const float* trans         = (const float*)d_in[3];
  const int* inc_idx         = (const int*)d_in[4];
  // d_in[5] inc_mask unused: inc_idx < 0 encodes it exactly.

  const int B = in_sizes[1] / TLEN;           // tags is (B, T)
  const int M = in_sizes[3] / KSTATES;        // transitions is (K, M)

  const size_t wsw_halves = (size_t)NT * KKT * 32 * 16;   // 295,936
  f16*   wsw  = (f16*)d_ws;                               // 578 KB swizzled W
  float* logZ = (float*)((char*)d_ws + wsw_halves * 2);   // 64 floats
  float* gold = logZ + 64;                                // 64 floats

  const int zero_u32 = (int)(wsw_halves / 2) + 128;
  zero_ws_kernel<<<256, 256, 0, stream>>>((uint32_t*)d_ws, zero_u32);
  build_w_kernel<<<(KSTATES * M + 255) / 256, 256, 0, stream>>>(
      trans, inc_idx, wsw, M);
  crf_forward_kernel<<<(B + 15) / 16, 1024, 0, stream>>>(
      em, mask, wsw, logZ, B);
  gold_kernel<<<B, 256, 0, stream>>>(em, tags, mask, trans, inc_idx, gold, M);
  final_kernel<<<1, 64, 0, stream>>>(logZ, gold, (float*)d_out, B);
}
